// NetworkOneNeuronLearnAll_34359738845
// MI455X (gfx1250) — matrix operand order repaired and checked by reading, not compiled, not run on hardware
//
#include <hip/hip_runtime.h>
#include <hip/hip_bf16.h>

typedef __attribute__((ext_vector_type(16))) _Float16 v16h;
typedef __attribute__((ext_vector_type(8)))  float    v8f;

#define Hh    64
#define T_    512
#define B_    4096
#define ROWS  16            // batch rows per block
#define NG    256           // 4*H gate columns
#define TB    (T_ * B_)

#define LOG2E 1.4426950408889634f
#define LN2   0.6931471805599453f

__device__ __forceinline__ float fast_exp(float v) {
#if __has_builtin(__builtin_amdgcn_exp2f)
  return __builtin_amdgcn_exp2f(LOG2E * v);
#else
  return expf(v);
#endif
}
__device__ __forceinline__ float fast_rcp(float v) {
#if __has_builtin(__builtin_amdgcn_rcpf)
  return __builtin_amdgcn_rcpf(v);
#else
  return 1.f / v;
#endif
}
__device__ __forceinline__ float fast_log(float v) {
#if __has_builtin(__builtin_amdgcn_logf)
  return LN2 * __builtin_amdgcn_logf(v);
#else
  return logf(v);
#endif
}
__device__ __forceinline__ float fast_tanh(float v) {
#if __has_builtin(__builtin_amdgcn_tanhf)
  return __builtin_amdgcn_tanhf(v);          // CDNA5 v_tanh_f32
#else
  return fmaf(-2.f, fast_rcp(1.f + fast_exp(2.f * v)), 1.f);
#endif
}
__device__ __forceinline__ float fast_softplus(float z) {   // log1p(exp(z))
  return fast_log(1.f + fast_exp(z));
}

// Persistent LSTM-scan: one block owns ROWS batch rows for all T steps.
// gates(16x256) = h(16x64) @ W_hh^T : 8 waves x 2 N-tiles x 2 K-slices of
// v_wmma_f32_16x16x32_f16 (W_hh resident in B-fragment VGPRs).
// gta(16x3) = h_t @ W_ll^T : wave 0 reuses the freshly packed A-fragments
// with a zero-padded W_ll B-fragment (2 more WMMAs).
// Activations are branchless: sigmoid(v) = 0.5*tanh(0.5*v)+0.5, so every
// gate is amul*v_tanh(ascale*v)+aoff with per-tile constants (3 VALU each).
__global__ __launch_bounds__(256)
void lstm_scan_wmma(const float* __restrict__ x,
                    const float* __restrict__ W_ih,   // (256,2)
                    const float* __restrict__ W_hh,   // (256,64)
                    const float* __restrict__ b_ih,   // (256)
                    const float* __restrict__ b_hh,   // (256)
                    const float* __restrict__ W_ll,   // (3,64)
                    const float* __restrict__ b_ll,   // (3)
                    const float* __restrict__ W_h1,   // (10,1)
                    const float* __restrict__ b_h1,   // (10)
                    const float* __restrict__ W_h2,   // (1,10)
                    const float* __restrict__ b_h2,   // (1)
                    float* __restrict__ out)          // [out|g|th|a] each T*B
{
  __shared__ __align__(32) _Float16 ldsA[2][2][32][16]; // h as A-fragments, ping-pong
  __shared__ float act[ROWS][NG];         // activated gates [m][n]
  __shared__ float x_lds[2][ROWS];        // ping-pong staged x_t
  __shared__ float out_buf[2][ROWS];      // prev-step output (feeds comb)
  __shared__ float gta_buf[2][3][ROWS];   // prev-step g/th/a (feeds softplus)

  const int tid     = threadIdx.x;
  const int lane    = tid & 31;
  const int wave    = tid >> 5;
  const int rowbase = blockIdx.x * ROWS;
  const int ncol    = lane & 15;

  // lin(x) is affine: lin = Ac*x + Cc
  float Ac = 0.f, Cc = 0.f;
  #pragma unroll
  for (int j = 0; j < 10; ++j) { Ac += W_h2[j] * W_h1[j]; Cc += W_h2[j] * b_h1[j]; }
  Cc += b_h2[0];

  // Per-lane gate-column constants + resident W_hh B-fragments (loaded once).
  float cb[2], w0[2], w1[2];
  float ascale[2], amul[2], aoff[2];      // branchless activation constants
  v16h bfrag[2][2];
  #pragma unroll
  for (int i = 0; i < 2; ++i) {
    const int n = (wave * 2 + i) * 16 + ncol;
    cb[i] = b_ih[n] + b_hh[n];
    w0[i] = W_ih[2 * n + 0];
    w1[i] = W_ih[2 * n + 1];
    const int gate = ((wave * 2 + i) * 16) >> 6;   // tile-uniform: 0:i 1:f 2:g 3:o
    ascale[i] = (gate == 2) ? 1.f : 0.5f;
    amul[i]   = (gate == 2) ? 1.f : 0.5f;
    aoff[i]   = (gate == 2) ? 0.f : 0.5f;
    #pragma unroll
    for (int s = 0; s < 2; ++s)
      #pragma unroll
      for (int j = 0; j < 16; ++j) {
        const int k = s * 32 + ((lane >= 16) ? 16 : 0) + j;
        bfrag[i][s][j] = (_Float16)W_hh[n * Hh + k];
      }
  }
  // Zero-padded W_ll B-fragment (columns 0..2 real, 3..15 zero) + bias
  v16h bllf[2];
  const float bll_r = (ncol < 3) ? b_ll[ncol] : 0.f;
  #pragma unroll
  for (int s = 0; s < 2; ++s)
    #pragma unroll
    for (int j = 0; j < 16; ++j) {
      const int k = s * 32 + ((lane >= 16) ? 16 : 0) + j;
      bllf[s][j] = (ncol < 3) ? (_Float16)W_ll[ncol * Hh + k] : (_Float16)0.f;
    }

  // t = 0 carry + outputs: h0 = c0 = 0, out0 = softplus(lin0 - 1), gta = 1
  for (int i = tid; i < 2 * 32 * 16; i += 256)
    ((_Float16*)ldsA[0])[i] = (_Float16)0.f;
  if (tid < ROWS) {
    const int b = rowbase + tid;
    const float o0 = fast_softplus(Ac * x[b] + Cc - 1.f);
    out[b]          = o0;
    out[TB + b]     = 1.f;
    out[2 * TB + b] = 1.f;
    out[3 * TB + b] = 1.f;
    out_buf[0][tid] = o0;
    gta_buf[0][0][tid] = 1.f;
    gta_buf[0][1][tid] = 1.f;
    gta_buf[0][2][tid] = 1.f;
    x_lds[1][tid] = x[B_ + b];           // stage x for t = 1
  }

  float creg[4] = {0.f, 0.f, 0.f, 0.f};  // c for this thread's 4 (m,u) cells

  for (int t = 1; t < T_; ++t) {
    const int q = t & 1;
    const int p = q ^ 1;
    __syncthreads();   // prev iteration's tail-phase writes visible

    // ---- phase 1: gates = bias + W_ih*[x, out_prev] + h_prev @ W_hh^T ----
    v8f acc[2];
    #pragma unroll
    for (int i = 0; i < 2; ++i)
      #pragma unroll
      for (int r = 0; r < 8; ++r) {
        const int m = r + ((lane < 16) ? 0 : 8);
        acc[i][r] = cb[i] + w0[i] * x_lds[q][m] + w1[i] * out_buf[p][m];
      }
    #pragma unroll
    for (int s = 0; s < 2; ++s) {
      const v16h a = *(const v16h*)(&ldsA[p][s][lane][0]);
      #pragma unroll
      for (int i = 0; i < 2; ++i)
        acc[i] = __builtin_amdgcn_wmma_f32_16x16x32_f16(
            false, a, false, bfrag[i][s], (short)0, acc[i], false, false);
    }
    // branchless activations -> LDS (v_mul + v_tanh + v_fma each)
    #pragma unroll
    for (int i = 0; i < 2; ++i) {
      const int n = (wave * 2 + i) * 16 + ncol;
      #pragma unroll
      for (int r = 0; r < 8; ++r) {
        const int m = r + ((lane < 16) ? 0 : 8);
        act[m][n] = fmaf(amul[i], fast_tanh(ascale[i] * acc[i][r]), aoff[i]);
      }
    }
    __syncthreads();

    // ---- phase 2: cell update; repack h_t into next A-fragments ----
    #pragma unroll
    for (int j = 0; j < 4; ++j) {
      const int e = tid + 256 * j;
      const int u = e & (Hh - 1);
      const int m = e >> 6;
      const float iv = act[m][u];
      const float fv = act[m][64 + u];
      const float gv = act[m][128 + u];
      const float ov = act[m][192 + u];
      creg[j] = fv * creg[j] + iv * gv;
      const float hv = ov * fast_tanh(creg[j]);
      const int s  = u >> 5;
      const int ku = u & 31;
      const int fl = m + ((ku >= 16) ? 16 : 0);
      const int fi = ku & 15;
      ldsA[q][s][fl][fi] = (_Float16)hv;
    }
    __syncthreads();

    // ---- phase 3 (tail): gta via WMMA on wave 0; emit outputs; stage x ----
    if (wave == 0) {                         // wave-uniform branch: EXEC all 1s
      v8f gacc = {};
      #pragma unroll
      for (int s = 0; s < 2; ++s) {
        const v16h a = *(const v16h*)(&ldsA[q][s][lane][0]);
        gacc = __builtin_amdgcn_wmma_f32_16x16x32_f16(
            false, a, false, bllf[s], (short)0, gacc, false, false);
      }
      if (ncol < 3) {
        #pragma unroll
        for (int r = 0; r < 8; ++r) {
          const int m = r + ((lane < 16) ? 0 : 8);
          const float v = gacc[r] + bll_r;
          gta_buf[q][ncol][m] = v;
          out[(1 + ncol) * TB + t * B_ + rowbase + m] = v;
        }
      }
    }
    if (tid < ROWS) {
      // adaptive softplus output (uses PREVIOUS step's g/th/a + current lin)
      const float lint = Ac * x_lds[q][tid] + Cc;
      const float pg  = gta_buf[p][0][tid];
      const float pth = gta_buf[p][1][tid];
      const float pa  = gta_buf[p][2][tid];
      const float onew = pg * fast_softplus(pa * (lint - pth)) * fast_rcp(pa);
      out_buf[q][tid] = onew;
      out[t * B_ + rowbase + tid] = onew;
      if (t + 1 < T_) {
        x_lds[p][tid] = x[(t + 1) * B_ + rowbase + tid];  // stage next x
        if (t + 2 < T_)
          __builtin_prefetch(&x[(t + 2) * B_ + rowbase + tid], 0, 0);
      }
    }
  }
}

extern "C" void kernel_launch(void* const* d_in, const int* in_sizes, int n_in,
                              void* d_out, int out_size, void* d_ws, size_t ws_size,
                              hipStream_t stream) {
  (void)in_sizes; (void)n_in; (void)out_size; (void)d_ws; (void)ws_size;
  lstm_scan_wmma<<<B_ / ROWS, 256, 0, stream>>>(
      (const float*)d_in[0],  (const float*)d_in[1], (const float*)d_in[2],
      (const float*)d_in[3],  (const float*)d_in[4], (const float*)d_in[5],
      (const float*)d_in[6],  (const float*)d_in[7], (const float*)d_in[8],
      (const float*)d_in[9],  (const float*)d_in[10], (float*)d_out);
}
